// SimpleCL_55490977465142
// MI455X (gfx1250) — compile-verified, matrix operand order checked by reading, not run on hardware
//
#include <hip/hip_runtime.h>

// ---------- types ----------
typedef __attribute__((ext_vector_type(16))) _Float16     v16h;
typedef __attribute__((ext_vector_type(8)))  float        v8f;
typedef __attribute__((ext_vector_type(8)))  unsigned int v8u;

#define NNODES 100000
#define NEDGES 1600000
#define C_IN   128
#define C_HID  256

static __device__ __forceinline__ unsigned pack2h(float x, float y) {
    _Float16 hx = (_Float16)x, hy = (_Float16)y;
    unsigned short ux = __builtin_bit_cast(unsigned short, hx);
    unsigned short uy = __builtin_bit_cast(unsigned short, hy);
    return (unsigned)ux | ((unsigned)uy << 16);
}

// ---------- utility kernels ----------
__global__ void k_zero4(float4* __restrict__ p, int n) {
    int i = blockIdx.x * blockDim.x + threadIdx.x;
    if (i < n) p[i] = make_float4(0.f, 0.f, 0.f, 0.f);
}

// Pack W [K,256] fp32 into WMMA B-fragment order:
//   Wpk[(((kblk*256 + col)*2 + hi)*8 + v] = pack2h(W[kg,col], W[kg+1,col])
//   kg = kblk*32 + 16*(v>=4) + 8*hi + 2*(v&3)
// so each lane's 8 fragment uints are 32 contiguous bytes.
__global__ void k_pack_w(const float* __restrict__ W, unsigned* __restrict__ Wpk, int K) {
    int idx = blockIdx.x * blockDim.x + threadIdx.x;
    int n = K * 128;                      // total packed uints = (K/32)*256*2*8
    if (idx >= n) return;
    int v    = idx & 7;
    int hi   = (idx >> 3) & 1;
    int c    = (idx >> 4) & 255;
    int kblk = idx >> 12;
    int kg   = kblk * 32 + ((v >> 2) << 4) + hi * 8 + ((v & 3) << 1);
    Wpk[idx] = pack2h(W[(size_t)kg * 256 + c], W[(size_t)(kg + 1) * 256 + c]);
}

__global__ void k_degree(const int* __restrict__ dst, float* __restrict__ deg, int E) {
    int i = blockIdx.x * blockDim.x + threadIdx.x;
    if (i < E) atomicAdd(&deg[dst[i]], 1.0f);
}

__global__ void k_invdeg(float* __restrict__ deg, int N) {
    int i = blockIdx.x * blockDim.x + threadIdx.x;
    if (i < N) deg[i] = 1.0f / fmaxf(deg[i], 1.0f);
}

// scatter-add feat[src[e]] into sums[dst[e]], F channels, 4 floats per thread
__global__ void k_scatter(const float* __restrict__ feat, const int* __restrict__ src,
                          const int* __restrict__ dst, float* __restrict__ sums,
                          int E, int F, int chunkShift) {
    int idx   = blockIdx.x * blockDim.x + threadIdx.x;
    int total = E << chunkShift;
    if (idx >= total) return;
    int e = idx >> chunkShift;
    int c = (idx - (e << chunkShift)) << 2;
    int s = src[e];
    int d = dst[e];
    float4 v = *(const float4*)(feat + (size_t)s * F + c);
    float* p = sums + (size_t)d * F + c;
    atomicAdd(p + 0, v.x);
    atomicAdd(p + 1, v.y);
    atomicAdd(p + 2, v.z);
    atomicAdd(p + 3, v.w);
}

// ---------- fused SAGE GEMM:  out = act( (sum/deg)@Wl + self@Wr + bias ) ----------
// Block = 256 thr (8 waves). Block tile: 64 rows x 32 cols. Wave tile: 16x16.
// A staged via LDS as packed f16 (2x ds_load_b128 per frag);
// B from fragment-packed weights (2x global_load_b128 per frag).
__global__ void __launch_bounds__(256)
k_sage_gemm(const float* __restrict__ Amean,        // [N,K] scatter sums
            const float* __restrict__ Aself,        // [N,K] self features
            const float* __restrict__ invdeg,       // [N]
            const unsigned* __restrict__ Wlp,       // packed [K/32,256,2,8] uints
            const unsigned* __restrict__ Wrp,       // packed
            const float* __restrict__ bias,         // [256]
            float* __restrict__ outp,               // [N,256]
            int N, int K, int relu) {
    __shared__ unsigned ldsA[64 * 16];              // 64 rows x 32 cols f16, packed pairs

    const int tid     = threadIdx.x;
    const int wave    = tid >> 5;
    const int lane    = tid & 31;
    const int rowTile = blockIdx.x * 64;
    const int colTile = blockIdx.y * 32;
    const int waveRow = wave & 3;                   // 16-row sub-tile
    const int waveCol = wave >> 2;                  // 16-col sub-tile
    const int col     = colTile + waveCol * 16 + (lane & 15);
    const int hi      = lane >> 4;                  // 0/1: lane half

    // staging role: thread covers float4 #sq and #sq+4 of row srow
    const int srow = tid >> 2;                      // 0..63
    const int sq   = tid & 3;
    const int grow = rowTile + srow;

    const int localRow = waveRow * 16 + (lane & 15);
    const int aBase    = localRow * 16 + hi * 4;    // uint index of frag low half

    v8f acc = {};

    for (int pass = 0; pass < 2; ++pass) {
        const float*    A  = pass ? Aself : Amean;
        const unsigned* Wp = pass ? Wrp   : Wlp;
        float s = 1.0f;
        if (pass == 0) s = (grow < N) ? invdeg[grow] : 0.0f;

        for (int k0 = 0; k0 < K; k0 += 32) {
            __syncthreads();                        // protect previous iteration's reads
            // ---- stage 64x32 fp32 -> packed f16 in LDS (coalesced float4 loads) ----
            #pragma unroll
            for (int qq = 0; qq < 2; ++qq) {
                int q = sq + qq * 4;                // float4 index 0..7 within row
                float4 v = make_float4(0.f, 0.f, 0.f, 0.f);
                if (grow < N)
                    v = *(const float4*)(A + (size_t)grow * K + k0 + q * 4);
                ldsA[srow * 16 + q * 2 + 0] = pack2h(v.x * s, v.y * s);
                ldsA[srow * 16 + q * 2 + 1] = pack2h(v.z * s, v.w * s);
            }
            __syncthreads();

            // ---- A fragment: two 16B LDS reads at ISA-layout offsets ----
            uint4 a0 = *(const uint4*)&ldsA[aBase];
            uint4 a1 = *(const uint4*)&ldsA[aBase + 8];
            v8u au;
            au[0] = a0.x; au[1] = a0.y; au[2] = a0.z; au[3] = a0.w;
            au[4] = a1.x; au[5] = a1.y; au[6] = a1.z; au[7] = a1.w;
            v16h af = __builtin_bit_cast(v16h, au);

            // ---- B fragment: one 32B contiguous per-lane load from packed weights ----
            const v8u* bp = (const v8u*)(Wp +
                (((size_t)(k0 >> 5) * 256 + col) * 2 + hi) * 8);
            v16h bf = __builtin_bit_cast(v16h, *bp);

            acc = __builtin_amdgcn_wmma_f32_16x16x32_f16(
                      /*neg_a=*/false, af, /*neg_b=*/false, bf,
                      /*c_mod=*/(short)0, acc, /*reuse_a=*/false, /*reuse_b=*/false);
        }
    }

    // ---- epilogue: bias (+ReLU); C/D layout: VGPR v -> row v + 8*(lane>=16) ----
    float bv = bias[col];
    #pragma unroll
    for (int v = 0; v < 8; ++v) {
        int row = rowTile + waveRow * 16 + v + hi * 8;
        if (row < N) {
            float val = acc[v] + bv;
            if (relu) val = fmaxf(val, 0.0f);
            outp[(size_t)row * 256 + col] = val;
        }
    }
}

// ---------- decode: out[e] = dot(z[a], z[b]) over 256 ch; one wave32 per edge ----------
__global__ void k_decode(const float* __restrict__ z, const int* __restrict__ edges,
                         float* __restrict__ out, int E) {
    int gid  = blockIdx.x * (blockDim.x >> 5) + (threadIdx.x >> 5);
    int lane = threadIdx.x & 31;
    if (gid >= E) return;
    int a = edges[gid * 2 + 0];
    int b = edges[gid * 2 + 1];
    const float4* za = (const float4*)(z + (size_t)a * 256);
    const float4* zb = (const float4*)(z + (size_t)b * 256);
    float acc = 0.0f;
    float4 va = za[lane],      vb = zb[lane];
    acc += va.x * vb.x + va.y * vb.y + va.z * vb.z + va.w * vb.w;
    va = za[lane + 32];        vb = zb[lane + 32];
    acc += va.x * vb.x + va.y * vb.y + va.z * vb.z + va.w * vb.w;
    #pragma unroll
    for (int msk = 16; msk; msk >>= 1) acc += __shfl_xor(acc, msk, 32);
    if (lane == 0) out[gid] = acc;
}

// ---------- host launch ----------
extern "C" void kernel_launch(void* const* d_in, const int* in_sizes, int n_in,
                              void* d_out, int out_size, void* d_ws, size_t ws_size,
                              hipStream_t stream) {
    (void)in_sizes; (void)n_in; (void)out_size; (void)ws_size;

    const float* x   = (const float*)d_in[0];
    const int*   ei  = (const int*)  d_in[1];   // [2,E]
    const int*   edg = (const int*)  d_in[2];   // [E,2]
    const float* W1l = (const float*)d_in[3];
    const float* b1  = (const float*)d_in[4];
    const float* W1r = (const float*)d_in[5];
    const float* W2l = (const float*)d_in[6];
    const float* b2  = (const float*)d_in[7];
    const float* W2r = (const float*)d_in[8];
    float* out = (float*)d_out;

    const int* srcI = ei;
    const int* dstI = ei + NEDGES;

    // workspace layout (256B aligned offsets)
    char* ws = (char*)d_ws;
    size_t off = 0;
    auto carve = [&](size_t bytes) -> char* {
        char* p = ws + off;
        off = (off + bytes + 255) & ~(size_t)255;
        return p;
    };
    float*    sum1  = (float*)   carve((size_t)NNODES * C_IN  * 4);  // 51.2 MB
    float*    sum2  = (float*)   carve((size_t)NNODES * C_HID * 4);  // 102.4 MB
    float*    deg   = (float*)   carve((size_t)NNODES * 4);          // 0.4 MB
    size_t zeroEnd  = off;                                           // zero [ws, ws+zeroEnd)
    float*    h     = (float*)   carve((size_t)NNODES * C_HID * 4);  // 102.4 MB
    float*    z     = (float*)   carve((size_t)NNODES * C_HID * 4);  // 102.4 MB
    unsigned* w1lp  = (unsigned*)carve((size_t)C_IN  * 512);         // 64 KB
    unsigned* w1rp  = (unsigned*)carve((size_t)C_IN  * 512);
    unsigned* w2lp  = (unsigned*)carve((size_t)C_HID * 512);         // 128 KB
    unsigned* w2rp  = (unsigned*)carve((size_t)C_HID * 512);

    const int BT = 256;

    // 1) zero accumulators + degree (must happen every call: atomics accumulate)
    {
        int n4 = (int)(zeroEnd / 16);
        k_zero4<<<(n4 + BT - 1) / BT, BT, 0, stream>>>((float4*)ws, n4);
    }
    // 2) pack weights into WMMA B-fragment order (f16 pairs)
    k_pack_w<<<(C_IN  * 128 + BT - 1) / BT, BT, 0, stream>>>(W1l, w1lp, C_IN);
    k_pack_w<<<(C_IN  * 128 + BT - 1) / BT, BT, 0, stream>>>(W1r, w1rp, C_IN);
    k_pack_w<<<(C_HID * 128 + BT - 1) / BT, BT, 0, stream>>>(W2l, w2lp, C_HID);
    k_pack_w<<<(C_HID * 128 + BT - 1) / BT, BT, 0, stream>>>(W2r, w2rp, C_HID);
    // 3) degrees -> inverse degrees
    k_degree<<<(NEDGES + BT - 1) / BT, BT, 0, stream>>>(dstI, deg, NEDGES);
    k_invdeg<<<(NNODES + BT - 1) / BT, BT, 0, stream>>>(deg, NNODES);
    // 4) scatter x -> sum1   (128 ch => 32 float4-chunks per edge)
    {
        long total = (long)NEDGES << 5;
        k_scatter<<<(unsigned)((total + BT - 1) / BT), BT, 0, stream>>>(
            x, srcI, dstI, sum1, NEDGES, C_IN, 5);
    }
    // 5) layer-1 fused WMMA GEMM: h = relu(mean1@W1l + x@W1r + b1)
    {
        dim3 grid((NNODES + 63) / 64, C_HID / 32);
        k_sage_gemm<<<grid, BT, 0, stream>>>(sum1, x, deg, w1lp, w1rp, b1, h,
                                             NNODES, C_IN, /*relu=*/1);
    }
    // 6) scatter h -> sum2   (256 ch => 64 float4-chunks per edge)
    {
        long total = (long)NEDGES << 6;
        k_scatter<<<(unsigned)((total + BT - 1) / BT), BT, 0, stream>>>(
            h, srcI, dstI, sum2, NEDGES, C_HID, 6);
    }
    // 7) layer-2 fused WMMA GEMM: z = mean2@W2l + h@W2r + b2
    {
        dim3 grid((NNODES + 63) / 64, C_HID / 32);
        k_sage_gemm<<<grid, BT, 0, stream>>>(sum2, h, deg, w2lp, w2rp, b2, z,
                                             NNODES, C_HID, /*relu=*/0);
    }
    // 8) decode: 8 edges per 256-thread block (one wave32 per edge)
    {
        int edgesPerBlock = BT / 32;
        k_decode<<<(NEDGES + edgesPerBlock - 1) / edgesPerBlock, BT, 0, stream>>>(
            z, edg, out, NEDGES);
    }
}
